// LocalSelfAttention_74388833567034
// MI455X (gfx1250) — matrix operand order repaired and checked by reading, not compiled, not run on hardware
//
#include <hip/hip_runtime.h>
#include <hip/hip_bf16.h>
#include <stdint.h>

#define HEADS 12
#define HD    64
#define L     2048
#define EMBED 768
#define QKVN  (3 * HEADS * HD)   // 2304
#define PADW  32
#define SPAN  80                  // 16 queries -> 80-key span
#define SPANP 96                  // padded to multiple of 32 for WMMA K

typedef __attribute__((ext_vector_type(16))) __bf16     v16bf;
typedef __attribute__((ext_vector_type(8)))  float       v8f;
typedef __attribute__((ext_vector_type(8)))  unsigned int v8u;

__device__ __forceinline__ unsigned short f2bf(float f) {
  union { float f; unsigned int u; } c; c.f = f;
  unsigned int u = c.u;
  unsigned int r = (u + 0x7FFFu + ((u >> 16) & 1u)) >> 16;  // RNE
  return (unsigned short)r;
}

__global__ void cvt_f32_bf16(const float* __restrict__ src,
                             unsigned short* __restrict__ dst, int n) {
  int i = blockIdx.x * blockDim.x + threadIdx.x;
  if (i < n) dst[i] = f2bf(src[i]);
}

// D[M,N] = A[M,K] * B[K,N], A/B bf16 row-major. K multiple of 64.
// F32OUT=1: fp32 output + bias (bias non-null). F32OUT=0: bf16 output.
// Block = 128 threads (4 waves); 64x64 output tile; K step 64.
template <bool F32OUT>
__global__ __launch_bounds__(128)
void gemm_bf16_wmma(const unsigned short* __restrict__ A,
                    const unsigned short* __restrict__ B,
                    unsigned short* __restrict__ Dbf,
                    float* __restrict__ Df,
                    const float* __restrict__ bias,
                    int M, int N, int K) {
  __shared__ __align__(16) unsigned short Ald[64 * 64];   // [m][k]
  __shared__ __align__(16) unsigned short BldT[64 * 64];  // [n][k] (transposed)
  const int tid  = threadIdx.x;
  const int lane = tid & 31;
  const int wave = tid >> 5;
  const int nl   = lane & 15;
  const int h    = lane >> 4;
  const int m0t  = blockIdx.y * 64;
  const int n0   = blockIdx.x * 64;

  v8f zero = {0, 0, 0, 0, 0, 0, 0, 0};
  v8f acc[4];
#pragma unroll
  for (int nt = 0; nt < 4; ++nt) acc[nt] = zero;

  for (int k0 = 0; k0 < K; k0 += 64) {
    // --- Stage A tile (64 rows x 64 halves) via async global->LDS copy. ---
    // 512 16B chunks, 128 threads x 4. ASYNCcnt-tracked, no VGPR round trip.
#pragma unroll
    for (int it = 0; it < 4; ++it) {
      int idx = tid + it * 128;
      int m = idx >> 3, cc = idx & 7;              // 8 chunks per row
      const unsigned short* g = A + (size_t)(m0t + m) * K + k0 + cc * 8;
      unsigned int ldsb = (unsigned int)(size_t)&Ald[m * 64 + cc * 8];
      asm volatile("global_load_async_to_lds_b128 %0, %1, off"
                   :: "v"(ldsb), "v"((unsigned long long)(size_t)g)
                   : "memory");
    }
    // --- Stage B tile (64 k-rows x 64 cols) transposed into [n][k]; pack two
    //     k-rows per 32-bit LDS store. 256 tasks, 128 threads x 2. ---
#pragma unroll
    for (int it = 0; it < 2; ++it) {
      int idx = tid + it * 128;
      int p = idx >> 3, cc = idx & 7;              // k-row pair p, n-chunk cc
      const unsigned short* src0 = B + (size_t)(k0 + 2 * p) * N + n0 + cc * 8;
      uint4 r0 = *(const uint4*)src0;
      uint4 r1 = *(const uint4*)(src0 + N);
      const unsigned short* h0 = (const unsigned short*)&r0;
      const unsigned short* h1 = (const unsigned short*)&r1;
#pragma unroll
      for (int e = 0; e < 8; ++e) {
        unsigned int packed = (unsigned int)h0[e] | ((unsigned int)h1[e] << 16);
        *(unsigned int*)&BldT[(cc * 8 + e) * 64 + 2 * p] = packed;
      }
    }
    // Prefetch next K tile while this one is consumed.
    if (k0 + 64 < K) {
      __builtin_prefetch(A + (size_t)(m0t + (tid >> 1)) * K + k0 + 64, 0, 0);
      __builtin_prefetch(B + (size_t)(k0 + 64 + (tid >> 2)) * N + n0, 0, 0);
    }
    asm volatile("s_wait_asynccnt 0" ::: "memory");
    __syncthreads();

    // --- A fragments (ISA 16-bit A layout), two K=32 chunks. ---
    v16bf af[2];
#pragma unroll
    for (int c = 0; c < 2; ++c) {
      v8u au;
#pragma unroll
      for (int v = 0; v < 8; ++v) {
        int kd = 32 * c + ((v < 4) ? 0 : 16) + 8 * h + 2 * (v & 3);
        au[v] = *(const unsigned int*)&Ald[(wave * 16 + nl) * 64 + kd];
      }
      af[c] = __builtin_bit_cast(v16bf, au);
    }
#pragma unroll
    for (int nt = 0; nt < 4; ++nt) {
#pragma unroll
      for (int c = 0; c < 2; ++c) {
        v8u bu;
#pragma unroll
        for (int v = 0; v < 8; ++v) {
          int kd = 32 * c + ((v < 4) ? 0 : 16) + 8 * h + 2 * (v & 3);
          bu[v] = *(const unsigned int*)&BldT[(nt * 16 + nl) * 64 + kd];
        }
        v16bf bf = __builtin_bit_cast(v16bf, bu);
        acc[nt] = __builtin_amdgcn_wmma_f32_16x16x32_bf16(
            false, af[c], false, bf, (short)0, acc[nt], false, false);
      }
    }
    __syncthreads();
  }

  // --- Straight-line epilogue (templated, no runtime branches). ---
  if (F32OUT) {
    float bv[4];
#pragma unroll
    for (int nt = 0; nt < 4; ++nt) bv[nt] = bias[n0 + nt * 16 + nl];
#pragma unroll
    for (int nt = 0; nt < 4; ++nt) {
#pragma unroll
      for (int r = 0; r < 8; ++r) {
        int row = m0t + wave * 16 + r + 8 * h;
        Df[(size_t)row * N + n0 + nt * 16 + nl] = acc[nt][r] + bv[nt];
      }
    }
  } else {
#pragma unroll
    for (int nt = 0; nt < 4; ++nt) {
#pragma unroll
      for (int r = 0; r < 8; ++r) {
        int row = m0t + wave * 16 + r + 8 * h;
        Dbf[(size_t)row * N + n0 + nt * 16 + nl] = f2bf(acc[nt][r]);
      }
    }
  }
}

// One wave per (16-query tile, head). qkv: bf16 [L][2304] rows = per-head q|k|v.
__global__ __launch_bounds__(32)
void local_attn_wmma(const unsigned short* __restrict__ qkv,
                     unsigned short* __restrict__ attn) {
  __shared__ __align__(16) unsigned short Kld[SPAN * HD];     // [w][kd]
  __shared__ __align__(16) unsigned short VldsT[HD * SPANP];  // [n2][w]
  __shared__ __align__(16) unsigned short Plds[16 * SPANP];   // [m][w]
  const int lane = threadIdx.x;
  const int nl = lane & 15;
  const int h  = lane >> 4;
  const int i0 = blockIdx.x * 16;
  const int hh = blockIdx.y;
  const int j0 = i0 - PADW;
  const int qb = hh * 3 * HD;
  const int kb = qb + HD;
  const int vb = qb + 2 * HD;
  v8f zero = {0, 0, 0, 0, 0, 0, 0, 0};

  // Stage K span into LDS row-major [w][kd]; branch-free zero-fill:
  // clamp the row index, load unconditionally, AND with validity mask.
  for (int idx = lane; idx < SPAN * 8; idx += 32) {
    int w = idx >> 3, cc = idx & 7;
    int j = j0 + w;
    int jc = min(max(j, 0), L - 1);
    unsigned int msk = (j >= 0 && j < L) ? 0xFFFFFFFFu : 0u;
    uint4 d = *(const uint4*)(qkv + (size_t)jc * QKVN + kb + cc * 8);
    d.x &= msk; d.y &= msk; d.z &= msk; d.w &= msk;
    *(uint4*)&Kld[w * HD + cc * 8] = d;
  }
  // Stage V span transposed into LDS [n2][w]; same branch-free zero-fill.
  for (int idx = lane; idx < SPANP * 8; idx += 32) {
    int w = idx >> 3, cc = idx & 7;
    int j = j0 + w;
    int jc = min(max(j, 0), L - 1);
    unsigned int msk = (w < SPAN && j >= 0 && j < L) ? 0xFFFFFFFFu : 0u;
    uint4 d = *(const uint4*)(qkv + (size_t)jc * QKVN + vb + cc * 8);
    d.x &= msk; d.y &= msk; d.z &= msk; d.w &= msk;
    const unsigned short* ph = (const unsigned short*)&d;
#pragma unroll
    for (int e = 0; e < 8; ++e) VldsT[(cc * 8 + e) * SPANP + w] = ph[e];
  }
  // Zero P pad columns (w = 80..95).
  for (int idx = lane; idx < 16 * 16; idx += 32)
    Plds[(idx >> 4) * SPANP + SPAN + (idx & 15)] = 0;

  // Q fragments (A layout), head-dim 64 = two K=32 chunks; direct global gathers.
  v16bf aq[2];
#pragma unroll
  for (int c = 0; c < 2; ++c) {
    v8u u;
#pragma unroll
    for (int v = 0; v < 8; ++v) {
      int kd = 32 * c + ((v < 4) ? 0 : 16) + 8 * h + 2 * (v & 3);
      u[v] = *(const unsigned int*)(qkv + (size_t)(i0 + nl) * QKVN + qb + kd);
    }
    aq[c] = __builtin_bit_cast(v16bf, u);
  }
  __syncthreads();

  // S(16x80) = Q * K^T : 5 n-tiles x 2 k-chunks of WMMA; unconditional
  // aligned ds_load_b32 fragment gathers (K pairs contiguous along head-dim).
  v8f s[5];
#pragma unroll
  for (int nt = 0; nt < 5; ++nt) {
    s[nt] = zero;
#pragma unroll
    for (int c = 0; c < 2; ++c) {
      v8u u;
#pragma unroll
      for (int v = 0; v < 8; ++v) {
        int kd = 32 * c + ((v < 4) ? 0 : 16) + 8 * h + 2 * (v & 3);
        u[v] = *(const unsigned int*)&Kld[(nt * 16 + nl) * HD + kd];
      }
      v16bf bk = __builtin_bit_cast(v16bf, u);
      s[nt] = __builtin_amdgcn_wmma_f32_16x16x32_bf16(
          false, aq[c], false, bk, (short)0, s[nt], false, false);
    }
  }

  // Mask (0 <= w-m <= 64 i.e. |key-query| <= 32) + softmax per row; rows live
  // in 16-lane halves, reduce with shfl_xor 1/2/4/8. Write P (bf16) to LDS.
#pragma unroll
  for (int r = 0; r < 8; ++r) {
    int m = r + 8 * h;
    float vals[5];
    float rmax = -1e30f;
#pragma unroll
    for (int nt = 0; nt < 5; ++nt) {
      int w = nt * 16 + nl;
      float sv = s[nt][r] * 0.125f;            // 1/sqrt(64)
      int d = w - m;
      if (d < 0 || d > 64) sv = -1e30f;
      vals[nt] = sv;
      rmax = fmaxf(rmax, sv);
    }
#pragma unroll
    for (int mk = 1; mk < 16; mk <<= 1)
      rmax = fmaxf(rmax, __shfl_xor(rmax, mk, 32));
    float sum = 0.0f;
#pragma unroll
    for (int nt = 0; nt < 5; ++nt) {
      float e = __expf(vals[nt] - rmax);
      vals[nt] = e;
      sum += e;
    }
#pragma unroll
    for (int mk = 1; mk < 16; mk <<= 1) sum += __shfl_xor(sum, mk, 32);
    float inv = 1.0f / sum;
#pragma unroll
    for (int nt = 0; nt < 5; ++nt)
      Plds[m * SPANP + nt * 16 + nl] = f2bf(vals[nt] * inv);
  }
  __syncthreads();

  // O(16x64) = P(16x96) * V(96x64): 3 k-chunks x 4 n-tiles.
  v16bf pa[3];
#pragma unroll
  for (int c = 0; c < 3; ++c) {
    v8u u;
#pragma unroll
    for (int v = 0; v < 8; ++v) {
      int w = 32 * c + ((v < 4) ? 0 : 16) + 8 * h + 2 * (v & 3);
      u[v] = *(const unsigned int*)&Plds[nl * SPANP + w];
    }
    pa[c] = __builtin_bit_cast(v16bf, u);
  }
  v8f o[4];
#pragma unroll
  for (int nt = 0; nt < 4; ++nt) {
    o[nt] = zero;
#pragma unroll
    for (int c = 0; c < 3; ++c) {
      v8u u;
#pragma unroll
      for (int v = 0; v < 8; ++v) {
        int w = 32 * c + ((v < 4) ? 0 : 16) + 8 * h + 2 * (v & 3);
        u[v] = *(const unsigned int*)&VldsT[(nt * 16 + nl) * SPANP + w];
      }
      v16bf bv = __builtin_bit_cast(v16bf, u);
      o[nt] = __builtin_amdgcn_wmma_f32_16x16x32_bf16(
          false, pa[c], false, bv, (short)0, o[nt], false, false);
    }
  }
  // Store O tile (bf16) into attn[L][768] at column hh*64.
#pragma unroll
  for (int nt = 0; nt < 4; ++nt) {
#pragma unroll
    for (int r = 0; r < 8; ++r) {
      int row = i0 + r + 8 * h;
      int col = hh * HD + nt * 16 + nl;
      attn[(size_t)row * EMBED + col] = f2bf(o[nt][r]);
    }
  }
}

extern "C" void kernel_launch(void* const* d_in, const int* in_sizes, int n_in,
                              void* d_out, int out_size, void* d_ws, size_t ws_size,
                              hipStream_t stream) {
  (void)in_sizes; (void)n_in; (void)out_size; (void)ws_size;
  const float* x    = (const float*)d_in[0];
  const float* wqkv = (const float*)d_in[1];
  const float* wout = (const float*)d_in[2];
  const float* bout = (const float*)d_in[3];
  float* out = (float*)d_out;

  // Workspace carve (~20.6 MB total): bf16 copies + intermediates.
  char* ws = (char*)d_ws;
  size_t off = 0;
  auto carve = [&](size_t bytes) -> unsigned short* {
    unsigned short* p = (unsigned short*)(ws + off);
    off += (bytes + 255) & ~(size_t)255;
    return p;
  };
  unsigned short* xb   = carve((size_t)L * EMBED * 2);
  unsigned short* wqb  = carve((size_t)EMBED * QKVN * 2);
  unsigned short* wob  = carve((size_t)EMBED * EMBED * 2);
  unsigned short* qkvb = carve((size_t)L * QKVN * 2);
  unsigned short* attb = carve((size_t)L * EMBED * 2);

  int n1 = L * EMBED, n2 = EMBED * QKVN, n3 = EMBED * EMBED;
  cvt_f32_bf16<<<(n1 + 255) / 256, 256, 0, stream>>>(x, xb, n1);
  cvt_f32_bf16<<<(n2 + 255) / 256, 256, 0, stream>>>(wqkv, wqb, n2);
  cvt_f32_bf16<<<(n3 + 255) / 256, 256, 0, stream>>>(wout, wob, n3);

  // QKV projection: [2048,768] x [768,2304] -> bf16 qkv.
  gemm_bf16_wmma<false><<<dim3(QKVN / 64, L / 64), 128, 0, stream>>>(
      xb, wqb, qkvb, nullptr, nullptr, L, QKVN, EMBED);

  // Windowed attention: one wave per (16-query tile, head).
  local_attn_wmma<<<dim3(L / 16, HEADS), 32, 0, stream>>>(qkvb, attb);

  // Output projection + bias: [2048,768] x [768,768] -> fp32 out.
  gemm_bf16_wmma<true><<<dim3(EMBED / 64, L / 64), 128, 0, stream>>>(
      attb, wob, nullptr, out, bout, L, EMBED, EMBED);
}
